// MultiHeadAttention_32306744001018
// MI455X (gfx1250) — compile-verified, matrix operand order checked by reading, not run on hardware
//
#include <hip/hip_runtime.h>
#include <hip/hip_bf16.h>

typedef unsigned short ushort_t;
typedef __attribute__((ext_vector_type(16))) __bf16 v16bf;
typedef __attribute__((ext_vector_type(8)))  float  v8f;

#define H_     16
#define HEAD_  64
#define S_     2048
#define B_     2
#define DMODEL 1024
#define MROWS  (B_ * S_)       // 4096
#define NQKV   (3 * DMODEL)    // 3072

// ---------- helpers ----------

__device__ __forceinline__ ushort_t f2bf(float f) {
  unsigned int u = __builtin_bit_cast(unsigned int, f);
  u += 0x7FFFu + ((u >> 16) & 1u);   // round-to-nearest-even
  return (ushort_t)(u >> 16);
}

// K-dim index held by (lane, element e) of a 16x32 bf16 A-fragment
// (ISA 7.12.2: lanes 0-15 hold K 0-7 / 16-23; lanes 16-31 hold K 8-15 / 24-31)
__device__ __forceinline__ int kmap(int lane, int e) {
  return (e & 7) + ((e >> 3) << 4) + ((lane >> 4) << 3);
}

union FragU { ushort_t u[16]; v16bf v; };

__device__ __forceinline__ v16bf load_frag(const ushort_t* p) {
  return *reinterpret_cast<const v16bf*>(p);   // 32B aligned by construction
}

__device__ __forceinline__ v8f wmma_bf16(v16bf a, v16bf b, v8f c) {
  return __builtin_amdgcn_wmma_f32_16x16x32_bf16(
      /*neg_a=*/false, a, /*neg_b=*/false, b,
      /*c_mod=*/(short)0, c, /*reuse_a=*/false, /*reuse_b=*/false);
}

// Intra-wave LDS ordering: same-wave DS ops are in-order (ISA 5.7.3); we only
// need the DScnt drained and a compiler barrier. No cross-wave sync required
// since each wave owns a private LDS slice.
__device__ __forceinline__ void lds_fence_wave() {
  asm volatile("s_wait_dscnt 0x0" ::: "memory");
}

// ---------- pack kernels ----------
// A-fragment pack (also valid as a B-fragment pack of W, since y = x @ W^T
// means B[k][n] = W[n][k]: pack W "rows" exactly like A rows).
// dst layout: [tile][kt][lane][16] contiguous (32B per lane per fragment).

__global__ __launch_bounds__(256) void pack_a_f32(const float* __restrict__ src,
                                                  ushort_t* __restrict__ dst,
                                                  int KT, int K) {
  int tid  = blockIdx.x * blockDim.x + threadIdx.x;
  int lane = tid & 31;
  int frag = tid >> 5;
  int kt   = frag % KT;
  int mt   = frag / KT;
  const float* s = src + (size_t)(mt * 16 + (lane & 15)) * K + kt * 32;
  ushort_t*    d = dst + ((size_t)frag * 32 + lane) * 16;
#pragma unroll
  for (int e = 0; e < 16; ++e) d[e] = f2bf(s[kmap(lane, e)]);
}

__global__ __launch_bounds__(256) void pack_a_bf16(const ushort_t* __restrict__ src,
                                                   ushort_t* __restrict__ dst,
                                                   int KT, int K) {
  int tid  = blockIdx.x * blockDim.x + threadIdx.x;
  int lane = tid & 31;
  int frag = tid >> 5;
  int kt   = frag % KT;
  int mt   = frag / KT;
  const ushort_t* s = src + (size_t)(mt * 16 + (lane & 15)) * K + kt * 32;
  ushort_t*       d = dst + ((size_t)frag * 32 + lane) * 16;
#pragma unroll
  for (int e = 0; e < 16; ++e) d[e] = s[kmap(lane, e)];
}

// Q (A-frag) and K^T (B-frag) per head share identical index math:
// lane&15 -> sequence position, kmap -> head-dim element.
// dst frag id = (((b*16+h)*128 + t16)*2 + dchunk)
__global__ __launch_bounds__(256) void pack_qk(const ushort_t* __restrict__ qkv,
                                               ushort_t* __restrict__ dst, int colBase) {
  int tid  = blockIdx.x * blockDim.x + threadIdx.x;
  int lane = tid & 31;
  int frag = tid >> 5;
  int dchunk = frag & 1;
  int rest   = frag >> 1;
  int t16    = rest & 127;  rest >>= 7;
  int h      = rest & 15;
  int b      = rest >> 4;
  int row  = b * S_ + t16 * 16 + (lane & 15);
  int colb = colBase + h * HEAD_ + dchunk * 32;
  const ushort_t* s = qkv + (size_t)row * NQKV + colb;
  ushort_t*       d = dst + ((size_t)frag * 32 + lane) * 16;
#pragma unroll
  for (int e = 0; e < 16; ++e) d[e] = s[kmap(lane, e)];
}

// V B-frag over (32 keys x 16 d): lane&15 -> d column, kmap -> key row.
// dst frag id = (((b*16+h)*64 + kt32)*4 + dt)
__global__ __launch_bounds__(256) void pack_v(const ushort_t* __restrict__ qkv,
                                              ushort_t* __restrict__ dst) {
  int tid  = blockIdx.x * blockDim.x + threadIdx.x;
  int lane = tid & 31;
  int frag = tid >> 5;
  int dt   = frag & 3;
  int rest = frag >> 2;
  int kt   = rest & 63;  rest >>= 6;
  int h    = rest & 15;
  int b    = rest >> 4;
  int col  = 2 * DMODEL + h * HEAD_ + dt * 16 + (lane & 15);
  ushort_t* d = dst + ((size_t)frag * 32 + lane) * 16;
#pragma unroll
  for (int e = 0; e < 16; ++e) {
    int row = b * S_ + kt * 32 + kmap(lane, e);
    d[e] = qkv[(size_t)row * NQKV + col];
  }
}

// ---------- WMMA GEMM: out[M,N] = A[M,K] @ W^T (+bias) ----------
// 128 threads = 4 waves arranged 2x2; block tile 128x64, wave tile 64x32
// (4 m-tiles x 2 n-tiles). Register double-buffering: loads for k+1 are in
// flight while 8 WMMAs consume the current buffer.
template <bool BF16OUT>
__global__ __launch_bounds__(128, 1) void gemm_ws(const ushort_t* __restrict__ Ap,
                                                  const ushort_t* __restrict__ Bp,
                                                  void* __restrict__ out,
                                                  const float* __restrict__ bias,
                                                  int KT, int ldc) {
  int lane = threadIdx.x & 31;
  int wave = threadIdx.x >> 5;
  int mtb  = blockIdx.y * 8 + (wave >> 1) * 4;   // 16-row tile index
  int ntb  = blockIdx.x * 4 + (wave & 1) * 2;    // 16-col tile index

  v8f acc[4][2] = {};

  const ushort_t* ap[4];
  const ushort_t* bp[2];
#pragma unroll
  for (int i = 0; i < 4; ++i) ap[i] = Ap + ((size_t)(mtb + i) * KT * 32 + lane) * 16;
#pragma unroll
  for (int j = 0; j < 2; ++j) bp[j] = Bp + ((size_t)(ntb + j) * KT * 32 + lane) * 16;

  v16bf a0[4], a1[4];
  v16bf b0[2], b1[2];
#pragma unroll
  for (int i = 0; i < 4; ++i) a0[i] = load_frag(ap[i]);
#pragma unroll
  for (int j = 0; j < 2; ++j) b0[j] = load_frag(bp[j]);

  for (int kt = 0; kt < KT; kt += 2) {
    // prefetch k-step kt+1 into buffer 1 (KT is even, so kt+1 < KT)
#pragma unroll
    for (int i = 0; i < 4; ++i) a1[i] = load_frag(ap[i] + 512);
#pragma unroll
    for (int j = 0; j < 2; ++j) b1[j] = load_frag(bp[j] + 512);
    // compute on buffer 0
#pragma unroll
    for (int i = 0; i < 4; ++i)
#pragma unroll
      for (int j = 0; j < 2; ++j)
        acc[i][j] = wmma_bf16(a0[i], b0[j], acc[i][j]);
    // prefetch k-step kt+2 into buffer 0
    if (kt + 2 < KT) {
#pragma unroll
      for (int i = 0; i < 4; ++i) a0[i] = load_frag(ap[i] + 1024);
#pragma unroll
      for (int j = 0; j < 2; ++j) b0[j] = load_frag(bp[j] + 1024);
    }
    // compute on buffer 1
#pragma unroll
    for (int i = 0; i < 4; ++i)
#pragma unroll
      for (int j = 0; j < 2; ++j)
        acc[i][j] = wmma_bf16(a1[i], b1[j], acc[i][j]);
#pragma unroll
    for (int i = 0; i < 4; ++i) ap[i] += 1024;
#pragma unroll
    for (int j = 0; j < 2; ++j) bp[j] += 1024;
  }

  int half = lane >> 4;
  int nl   = lane & 15;
#pragma unroll
  for (int i = 0; i < 4; ++i)
#pragma unroll
    for (int j = 0; j < 2; ++j) {
      int colb = (ntb + j) * 16 + nl;
#pragma unroll
      for (int g = 0; g < 8; ++g) {
        int row = (mtb + i) * 16 + g + half * 8;
        float v = acc[i][j][g];
        if constexpr (BF16OUT) {
          ((ushort_t*)out)[(size_t)row * ldc + colb] = f2bf(v);
        } else {
          ((float*)out)[(size_t)row * ldc + colb] = v + bias[colb];
        }
      }
    }
}

// ---------- flash attention (per-head, causal, WMMA) ----------
// Block = 4 waves; each wave independently owns one 16-row q tile and a
// private 1KB LDS slice (no cross-wave barriers). Causal structure exploited:
// each wave runs only kt <= qt>>1 chunks, and the mask is evaluated only on
// the final (diagonal) chunk. All K/V fragments for a chunk are issued up
// front so V loads hide under the softmax VALU block.
__global__ __launch_bounds__(128, 1) void flash_attn(const ushort_t* __restrict__ Qp,
                                                     const ushort_t* __restrict__ Kp,
                                                     const ushort_t* __restrict__ Vp,
                                                     ushort_t* __restrict__ ctx) {
  __shared__ ushort_t lds[4][16 * 32];

  int lane = threadIdx.x & 31;
  int wave = threadIdx.x >> 5;
  int qwg  = blockIdx.x & 31;              // 64-row workgroup q tile
  int h    = (blockIdx.x >> 5) & 15;
  int b    = blockIdx.x >> 9;
  int bh   = b * H_ + h;
  int qt   = qwg * 4 + wave;               // 16-row q tile (0..127)
  int qbase = qt * 16;
  int half = lane >> 4;
  int nl   = lane & 15;

  v16bf aQ0, aQ1;
  {
    const ushort_t* qp = Qp + (((size_t)bh * 128 + qt) * 2 * 32 + lane) * 16;
    aQ0 = load_frag(qp);
    aQ1 = load_frag(qp + 512);
  }

  float m[8], l[8];
#pragma unroll
  for (int g = 0; g < 8; ++g) { m[g] = -1e30f; l[g] = 0.f; }
  v8f acc[4] = {};

  const float inv_scale = 0.17677669529663687f;  // 1 / 1024^0.25 (faithful quirk)
  int ktLast = qt >> 1;   // last chunk containing any key <= any of our q rows

  const ushort_t* kp = Kp + (((size_t)bh * 128) * 2 * 32 + lane) * 16;  // +2048/chunk
  const ushort_t* vp = Vp + (((size_t)bh * 64) * 4 * 32 + lane) * 16;   // +2048/chunk

  for (int kt = 0; kt <= ktLast; ++kt, kp += 2048, vp += 2048) {
    // --- issue ALL loads for this 32-key chunk up front ---
    v16bf bK[4];
#pragma unroll
    for (int j = 0; j < 4; ++j) bK[j] = load_frag(kp + j * 512);  // [sub0 d0|d1, sub1 d0|d1]
    v16bf bV[4];
#pragma unroll
    for (int dt = 0; dt < 4; ++dt) bV[dt] = load_frag(vp + dt * 512);

    // --- scores: two 16-key subtiles, K-dim = head dim (64 = 2 chunks) ---
    v8f s[2];
#pragma unroll
    for (int j = 0; j < 2; ++j) {
      v8f t = {};
      t = wmma_bf16(aQ0, bK[2 * j + 0], t);
      t = wmma_bf16(aQ1, bK[2 * j + 1], t);
      s[j] = t;
    }

    // --- scale (+ causal mask only on the diagonal chunk) ---
#pragma unroll
    for (int j = 0; j < 2; ++j)
#pragma unroll
      for (int g = 0; g < 8; ++g) s[j][g] *= inv_scale;
    if (kt == ktLast) {
      int kbase = kt * 32;
#pragma unroll
      for (int j = 0; j < 2; ++j) {
        int kcol = kbase + j * 16 + nl;
#pragma unroll
        for (int g = 0; g < 8; ++g) {
          int qrow = qbase + g + half * 8;
          if (kcol > qrow) s[j][g] = -1e30f;
        }
      }
    }

    // --- online softmax: row stats via 16-lane-half shfl reductions ---
    float scale_f[8];
#pragma unroll
    for (int g = 0; g < 8; ++g) {
      float t = fmaxf(s[0][g], s[1][g]);
#pragma unroll
      for (int msk = 1; msk < 16; msk <<= 1) t = fmaxf(t, __shfl_xor(t, msk, 32));
      float mn = fmaxf(m[g], t);
      scale_f[g] = __expf(m[g] - mn);
      m[g] = mn;
      float p0 = __expf(s[0][g] - mn);
      float p1 = __expf(s[1][g] - mn);
      s[0][g] = p0;  s[1][g] = p1;
      float rs = p0 + p1;
#pragma unroll
      for (int msk = 1; msk < 16; msk <<= 1) rs += __shfl_xor(rs, msk, 32);
      l[g] = l[g] * scale_f[g] + rs;
    }

    // --- rescale running context ---
#pragma unroll
    for (int dt = 0; dt < 4; ++dt)
#pragma unroll
      for (int g = 0; g < 8; ++g) acc[dt][g] = acc[dt][g] * scale_f[g];

    // --- P (C/D layout) -> private LDS slice -> A-fragment (wave-local) ---
#pragma unroll
    for (int g = 0; g < 8; ++g) {
      int row = g + half * 8;
      lds[wave][row * 32 + nl]      = f2bf(s[0][g]);
      lds[wave][row * 32 + 16 + nl] = f2bf(s[1][g]);
    }
    lds_fence_wave();
    FragU fp;
    {
      int row = nl;
      int kb  = half * 8;
#pragma unroll
      for (int e = 0; e < 8; ++e) {
        fp.u[e]     = lds[wave][row * 32 + kb + e];
        fp.u[e + 8] = lds[wave][row * 32 + kb + 16 + e];
      }
    }
    lds_fence_wave();

    // --- ctx += P @ V (4 d-tiles of 16), V already resident ---
#pragma unroll
    for (int dt = 0; dt < 4; ++dt)
      acc[dt] = wmma_bf16(fp.v, bV[dt], acc[dt]);
  }

  // --- finalize: divide by l, store bf16 ctx row-major [B*S, 1024] ---
#pragma unroll
  for (int g = 0; g < 8; ++g) l[g] = 1.0f / l[g];
#pragma unroll
  for (int dt = 0; dt < 4; ++dt) {
    int col = h * HEAD_ + dt * 16 + nl;
#pragma unroll
    for (int g = 0; g < 8; ++g) {
      int row = b * S_ + qbase + g + half * 8;
      ctx[(size_t)row * DMODEL + col] = f2bf(acc[dt][g] * l[g]);
    }
  }
}

// ---------- host launcher ----------

extern "C" void kernel_launch(void* const* d_in, const int* in_sizes, int n_in,
                              void* d_out, int out_size, void* d_ws, size_t ws_size,
                              hipStream_t stream) {
  (void)in_sizes; (void)n_in; (void)out_size; (void)ws_size;
  const float* x  = (const float*)d_in[0];
  const float* Wq = (const float*)d_in[1];
  const float* Wk = (const float*)d_in[2];
  const float* Wv = (const float*)d_in[3];
  const float* Wo = (const float*)d_in[4];
  const float* bo = (const float*)d_in[5];

  char* ws = (char*)d_ws;
  size_t off = 0;
  auto take = [&](size_t bytes) -> char* {
    char* p = ws + off;
    off += (bytes + 255) & ~(size_t)255;
    return p;
  };
  ushort_t* xp    = (ushort_t*)take((size_t)MROWS * DMODEL * 2);   // x A-frags
  ushort_t* wqkvp = (ushort_t*)take((size_t)NQKV  * DMODEL * 2);   // Wq|Wk|Wv B-frags
  ushort_t* wop   = (ushort_t*)take((size_t)DMODEL * DMODEL * 2);  // Wo B-frags
  ushort_t* qkv   = (ushort_t*)take((size_t)MROWS * NQKV * 2);     // fused qkv bf16
  ushort_t* Qp    = (ushort_t*)take((size_t)B_ * H_ * S_ * HEAD_ * 2);
  ushort_t* Kp    = (ushort_t*)take((size_t)B_ * H_ * S_ * HEAD_ * 2);
  ushort_t* Vp    = (ushort_t*)take((size_t)B_ * H_ * S_ * HEAD_ * 2);
  ushort_t* ctx   = (ushort_t*)take((size_t)MROWS * DMODEL * 2);
  ushort_t* ctxp  = (ushort_t*)take((size_t)MROWS * DMODEL * 2);

  const int KT = DMODEL / 32;  // 32 k-steps

  // 1) pack x into A-fragments
  {
    int frags = (MROWS / 16) * KT;               // 8192
    pack_a_f32<<<dim3(frags / 8), dim3(256), 0, stream>>>(x, xp, KT, DMODEL);
  }
  // 2) pack weights into B-fragments (A-pack of W == B-frag of W^T)
  {
    int frags  = (DMODEL / 16) * KT;             // 2048
    int blocks = frags / 8;
    size_t sec = (size_t)(DMODEL / 16) * KT * 512;  // ushorts per weight section
    pack_a_f32<<<dim3(blocks), dim3(256), 0, stream>>>(Wq, wqkvp,           KT, DMODEL);
    pack_a_f32<<<dim3(blocks), dim3(256), 0, stream>>>(Wk, wqkvp + sec,     KT, DMODEL);
    pack_a_f32<<<dim3(blocks), dim3(256), 0, stream>>>(Wv, wqkvp + sec * 2, KT, DMODEL);
    pack_a_f32<<<dim3(blocks), dim3(256), 0, stream>>>(Wo, wop,             KT, DMODEL);
  }
  // 3) fused QKV projection: qkv = x @ [Wq|Wk|Wv]^T  (bf16 out)
  //    block tile 128x64 -> grid (N/64, M/128)
  gemm_ws<true><<<dim3(NQKV / 64, MROWS / 128), dim3(128), 0, stream>>>(
      xp, wqkvp, qkv, (const float*)nullptr, KT, NQKV);
  // 4) per-head fragment repack
  pack_qk<<<dim3(1024), dim3(256), 0, stream>>>(qkv, Qp, 0);
  pack_qk<<<dim3(1024), dim3(256), 0, stream>>>(qkv, Kp, DMODEL);
  pack_v <<<dim3(1024), dim3(256), 0, stream>>>(qkv, Vp);
  // 5) causal flash attention
  flash_attn<<<dim3(B_ * H_ * (S_ / 64)), dim3(128), 0, stream>>>(Qp, Kp, Vp, ctx);
  // 6) pack ctx into A-fragments
  {
    int frags = (MROWS / 16) * KT;
    pack_a_bf16<<<dim3(frags / 8), dim3(256), 0, stream>>>(ctx, ctxp, KT, DMODEL);
  }
  // 7) output projection with bias (fp32 into d_out)
  gemm_ws<false><<<dim3(DMODEL / 64, MROWS / 128), dim3(128), 0, stream>>>(
      ctxp, wop, d_out, bo, KT, DMODEL);
}
